// LSTM_Model_884763263105
// MI455X (gfx1250) — matrix operand order repaired and checked by reading, not compiled, not run on hardware
//
#include <hip/hip_runtime.h>
#include <hip/hip_bf16.h>

// ---------------------------------------------------------------------------
// LSTM (B=4096, T=1024, D=4, H=16) + FC(16->1) + tanh, for MI455X (gfx1250).
//
// Per wave: one tile of 16 batch rows. Per timestep, per gate g in {i,f,g,o}:
//   D_g[m=hid][n=batch] = A_g (16x32 f16, const) x B (32x16 f16) + 0
// K augmentation: K0..15 = h, K16..19 = x, K20 = 1.0 (bias lane), K21..31 = 0.
// A_g packs [w_hh_g | w_ih_g | bias_g | 0] along K -> bias folded into WMMA,
// C is the zero tile (no per-step accumulator copies).
// One v_wmma_f32_16x16x32_f16 per gate -> 4 WMMA / step / wave.
// Gate nonlinearities use v_tanh_f32 (1 TRANS op each).
// ---------------------------------------------------------------------------

typedef __attribute__((ext_vector_type(16))) _Float16 v16h;
typedef __attribute__((ext_vector_type(8)))  float    v8f;
typedef __attribute__((ext_vector_type(2)))  __fp16   fp16x2;

#define LSTM_B 4096
#define LSTM_T 1024
#define LSTM_D 4
#define LSTM_H 16

__device__ __forceinline__ float fast_tanh(float x) {
#if __has_builtin(__builtin_amdgcn_tanhf)
    return __builtin_amdgcn_tanhf(x);
#else
    float r;
    asm("v_tanh_f32 %0, %1\n\tv_nop" : "=v"(r) : "v"(x));
    return r;
#endif
}
__device__ __forceinline__ float fast_sigm(float x) {
    return __builtin_fmaf(0.5f, fast_tanh(0.5f * x), 0.5f);
}

// pack two f32 -> one dword of two f16 (v_cvt_pk_f16_f32 / pkrtz)
__device__ __forceinline__ unsigned pkh(float a, float b) {
    union { fp16x2 h; unsigned u; } t;
    t.h = __builtin_amdgcn_cvt_pkrtz(a, b);
    return t.u;
}

__device__ __forceinline__ v8f wmma_f16(v16h a, v16h b, v8f c) {
    // 8 args: (neg_a, A, neg_b, B, c_mod, C, reuse_a, reuse_b)
    return __builtin_amdgcn_wmma_f32_16x16x32_f16(
        false, a, false, b, (short)0, c, false, false);
}

__global__ __launch_bounds__(128, 1) void lstm_wmma_kernel(
    const float* __restrict__ x,     // [B,T,D]
    const float* __restrict__ w_ih,  // [4H,D]
    const float* __restrict__ w_hh,  // [4H,H]
    const float* __restrict__ b_ih,  // [4H]
    const float* __restrict__ b_hh,  // [4H]
    const float* __restrict__ w_fc,  // [1,H]
    const float* __restrict__ b_fc,  // [1]
    float* __restrict__ out)         // [B,1]
{
    const int lane = (int)(threadIdx.x & 31u);
    const int wave = (int)blockIdx.x * ((int)blockDim.x >> 5) + ((int)threadIdx.x >> 5);
    const int b0 = wave * 16;
    if (b0 >= LSTM_B) return;

    const int  m  = lane & 15;      // A: row M; B/D: batch column N
    const bool hi = lane >= 16;     // upper half-wave

    // ---- Constant A tiles (16x32 f16, one per gate). A[m][k]:
    //   k <16    : w_hh[g*16+m][k]
    //   k 16..19 : w_ih[g*16+m][k-16]
    //   k 20     : b_ih[g*16+m] + b_hh[g*16+m]   (pairs with B's 1.0 at K=20)
    //   k 21..31 : 0
    // 16-bit A 16x32 layout: lanes 0-15 hold K=0..7 (halves 0..7) and
    // K=16..23 (halves 8..15); lanes 16-31 hold K=8..15 and K=24..31.
    v16h A[4];
    #pragma unroll
    for (int g = 0; g < 4; ++g) {
        const int row = g * 16 + m;
        const int k0  = hi ? 8 : 0;
        v16h a = {};
        #pragma unroll
        for (int j = 0; j < 8; ++j)
            a[j] = (_Float16)w_hh[row * LSTM_H + k0 + j];
        if (!hi) {
            #pragma unroll
            for (int j = 0; j < LSTM_D; ++j)
                a[8 + j] = (_Float16)w_ih[row * LSTM_D + j];
            a[12] = (_Float16)(b_ih[row] + b_hh[row]);   // K=20
        }
        A[g] = a;
    }

    // ---- Recurrent state, f32 in D layout: hD[v]/cD[v] = state[hid][batch],
    // hid = v + 8*hi, batch = b0 + m.
    float hD[8], cD[8];
    #pragma unroll
    for (int v = 0; v < 8; ++v) { hD[v] = 0.0f; cD[v] = 0.0f; }

    // Each lane streams its own batch row of x: contiguous 16B per step.
    const float4* __restrict__ xrow =
        (const float4*)x + (size_t)(b0 + m) * LSTM_T;

    const v8f zc = {};   // zero C tile

    for (int t = 0; t < LSTM_T; ++t) {
        const float4 xv = xrow[t];
        {   // software prefetch ahead of the x stream
            const int tp = (t + 16 < LSTM_T) ? t + 16 : LSTM_T - 1;
            __builtin_prefetch((const void*)(xrow + tp), 0, 0);
        }

        // Pack h pairs to f16 dwords, then cross-half shuffle the packed
        // dwords (4 DS ops instead of 8).
        unsigned ph0 = pkh(hD[0], hD[1]);
        unsigned ph1 = pkh(hD[2], hD[3]);
        unsigned ph2 = pkh(hD[4], hD[5]);
        unsigned ph3 = pkh(hD[6], hD[7]);
        unsigned sh0 = (unsigned)__shfl_xor((int)ph0, 16, 32);
        unsigned sh1 = (unsigned)__shfl_xor((int)ph1, 16, 32);
        unsigned sh2 = (unsigned)__shfl_xor((int)ph2, 16, 32);
        unsigned sh3 = (unsigned)__shfl_xor((int)ph3, 16, 32);

        // ---- Build B tile (32x16 f16).
        // Lanes 0-15 : K=0..7 = h[batch][0..7], K=16..19 = x[batch][0..3], K=20 = 1.0, K=21..23 = 0
        // Lanes 16-31: K=8..15 = h[batch][8..15], K=24..31 = 0
        union { v16h v; unsigned u[8]; } Bm;
        Bm.u[0] = ph0;
        Bm.u[1] = ph1;
        Bm.u[2] = ph2;
        Bm.u[3] = ph3;
        Bm.u[4] = hi ? 0u : pkh(xv.x, xv.y);
        Bm.u[5] = hi ? 0u : pkh(xv.z, xv.w);
        Bm.u[6] = hi ? 0u : 0x00003C00u;   // halves K=20 (1.0h), K=21 (0)
        Bm.u[7] = 0u;

        // ---- 4 gate GEMMs: D_g = A_g x B (+bias via K=20). Order i,f,g,o.
        v8f Di = wmma_f16(A[0], Bm.v, zc);
        v8f Df = wmma_f16(A[1], Bm.v, zc);
        v8f Dg = wmma_f16(A[2], Bm.v, zc);
        v8f Do = wmma_f16(A[3], Bm.v, zc);

        // ---- Elementwise LSTM cell update (same D layout for all tiles).
        #pragma unroll
        for (int v = 0; v < 8; ++v) {
            const float cv = fast_sigm(Df[v]) * cD[v]
                           + fast_sigm(Di[v]) * fast_tanh(Dg[v]);
            cD[v] = cv;
            hD[v] = fast_sigm(Do[v]) * fast_tanh(cv);
        }
    }

    // ---- Final FC: out[b] = tanh( sum_hid h[hid][b]*w_fc[hid] + b_fc ).
    float part = 0.0f;
    #pragma unroll
    for (int v = 0; v < 8; ++v)
        part += hD[v] * w_fc[v + (hi ? 8 : 0)];
    part += __shfl_xor(part, 16, 32);   // combine hid 0..7 with 8..15
    if (!hi)
        out[b0 + m] = fast_tanh(part + b_fc[0]);
}

extern "C" void kernel_launch(void* const* d_in, const int* in_sizes, int n_in,
                              void* d_out, int out_size, void* d_ws, size_t ws_size,
                              hipStream_t stream) {
    (void)in_sizes; (void)n_in; (void)d_ws; (void)ws_size; (void)out_size;
    const float* x    = (const float*)d_in[0];
    const float* w_ih = (const float*)d_in[1];
    const float* w_hh = (const float*)d_in[2];
    const float* b_ih = (const float*)d_in[3];
    const float* b_hh = (const float*)d_in[4];
    const float* w_fc = (const float*)d_in[5];
    const float* b_fc = (const float*)d_in[6];
    float* out = (float*)d_out;

    // 256 wave-tiles of 16 batch rows; 4 waves per block.
    const int threads = 128;
    const int blocks  = (LSTM_B / 16) / (threads / 32);  // 64
    lstm_wmma_kernel<<<blocks, threads, 0, stream>>>(
        x, w_ih, w_hh, b_ih, b_hh, w_fc, b_fc, out);
}
